// _YatPerformerPolyBase_16896401342692
// MI455X (gfx1250) — matrix operand order repaired and checked by reading, not compiled, not run on hardware
//
#include <hip/hip_runtime.h>
#include <hip/hip_bf16.h>
#include <math.h>
#include <stdint.h>

typedef __attribute__((ext_vector_type(16))) __bf16 v16bf;
typedef __attribute__((ext_vector_type(8)))  __bf16 v8bf;
typedef __attribute__((ext_vector_type(8)))  float  v8f;
typedef __attribute__((ext_vector_type(4)))  unsigned int v4u;
typedef __attribute__((ext_vector_type(8)))  int    v8i;
typedef __attribute__((ext_vector_type(4)))  int    v4i;

#define EMBED 512
#define T_LEN 1024
#define NH    8
#define HD    64          // head dim
#define PP    32          // poly_dim
#define MMF   4           // num_prf_features
#define FD    128         // feature dim = PP*MMF
#define CH    32          // chunk length (tokens per recurrence step)

#if defined(__HIP_DEVICE_COMPILE__) && __has_builtin(__builtin_amdgcn_tensor_load_to_lds)
#define HAVE_TDM 1
#else
#define HAVE_TDM 0
#endif

// ---------------------------------------------------------------------------
// WMMA helpers (CDNA5 gfx1250, wave32). Layouts used by this code (A/B in contiguous K order):
//  A (16x32 bf16): lane r=lane&15 holds row r; elem e -> K = (lane>>4)*16 + e
//  B (32x16 bf16): lane n=lane&15 holds col n; elem e -> K = (lane>>4)*16 + e
//  C/D (16x16 f32): elem i at lane L -> row = i + (L>>4)*8, col = L&15
// ---------------------------------------------------------------------------
static __device__ inline v8f wmma_bf16(v16bf a, v16bf b, v8f c) {
  return __builtin_amdgcn_wmma_f32_16x16x32_bf16(false, a, false, b, (short)0, c,
                                                 false, false);
}

// Row-major A fragment: two 128-bit LDS loads per lane (contiguous K runs).
static __device__ inline v16bf load_a_rm(const __bf16* M, int ld, int row0, int k0,
                                         int lane) {
  const int r = row0 + (lane & 15);
  const int base = (lane >> 4) << 4;
  const v8bf lo = *(const v8bf*)(M + r * ld + k0 + base);
  const v8bf hi = *(const v8bf*)(M + r * ld + k0 + 8 + base);
  return __builtin_shufflevector(lo, hi, 0, 1, 2, 3, 4, 5, 6, 7, 8, 9, 10, 11, 12,
                                 13, 14, 15);
}

// A = M^T  (M row-major [K x rows]):  A[r][k] = M[k][row0+r]  (strided)
static __device__ inline v16bf load_a_tr(const __bf16* M, int ld, int row0, int k0,
                                         int lane) {
  const int r = row0 + (lane & 15);
  const int base = (lane >> 4) << 4;
  v16bf a;
#pragma unroll
  for (int e = 0; e < 16; ++e) {
    const int kk = k0 + base + e;
    a[e] = M[kk * ld + r];
  }
  return a;
}

static __device__ inline v16bf load_b_rm(const __bf16* M, int ld, int k0, int col0,
                                         int lane) {
  const int n = col0 + (lane & 15);
  const int koff = k0 + ((lane >> 4) << 4);
  v16bf b;
#pragma unroll
  for (int e = 0; e < 16; ++e) b[e] = M[(koff + e) * ld + n];
  return b;
}

// B fragment from fp32 LDS with on-the-fly bf16 conversion (column read).
static __device__ inline v16bf load_b_rm_f32(const float* M, int ld, int k0,
                                             int col0, int lane) {
  const int n = col0 + (lane & 15);
  const int koff = k0 + ((lane >> 4) << 4);
  v16bf b;
#pragma unroll
  for (int e = 0; e < 16; ++e) b[e] = (__bf16)M[(koff + e) * ld + n];
  return b;
}

// B = M^T  (M row-major [N x K]): per-lane contiguous in k -> 2x b128 loads
static __device__ inline v16bf load_b_tr(const __bf16* M, int ld, int k0, int col0,
                                         int lane) {
  const int n = col0 + (lane & 15);
  const int koff = k0 + ((lane >> 4) << 4);
  const v8bf lo = *(const v8bf*)(M + n * ld + koff);
  const v8bf hi = *(const v8bf*)(M + n * ld + koff + 8);
  return __builtin_shufflevector(lo, hi, 0, 1, 2, 3, 4, 5, 6, 7, 8, 9, 10, 11, 12,
                                 13, 14, 15);
}

#if HAVE_TDM
// ---------------------------------------------------------------------------
// Tensor Data Mover: 2D f32 tile (tile_h rows x tile_w elems, row stride in
// elems) global -> LDS, contiguous in LDS. D# built per ISA ch.8:
//  g0: count=1 | lds_addr | global_addr | type=2
//  g1: data_size=4B, tensor_dim0/1, tile_dim0/1, tensor_dim0_stride
// ---------------------------------------------------------------------------
static __device__ inline void tdm_load_2d_f32(unsigned long long gaddr,
                                              unsigned lds_addr, unsigned tile_w,
                                              unsigned tile_h,
                                              unsigned long long stride_elems) {
  v4u g0;
  g0[0] = 1u;                                        // count=1, no gather
  g0[1] = lds_addr;                                  // bits 63:32
  g0[2] = (unsigned)(gaddr & 0xffffffffu);           // bits 95:64
  g0[3] = (unsigned)((gaddr >> 32) & 0x01ffffffu) | (2u << 30);  // addr hi | type=2
  v8i g1;
  g1[0] = (int)(2u << 16);                                        // data_size=4B
  g1[1] = (int)((tile_w & 0xffffu) << 16);                        // tensor_dim0 lo
  g1[2] = (int)(((tile_w >> 16) & 0xffffu) | ((tile_h & 0xffffu) << 16));
  g1[3] = (int)(((tile_h >> 16) & 0xffffu) | ((tile_w & 0xffffu) << 16)); // tile_dim0
  g1[4] = (int)(tile_h & 0xffffu);                                // tile_dim1
  g1[5] = (int)(unsigned)(stride_elems & 0xffffffffu);            // dim0 stride lo
  g1[6] = (int)(unsigned)((stride_elems >> 32) & 0xffffu);        // dim0 stride hi
  g1[7] = 0;
  v4i g2 = {0, 0, 0, 0};
  v4i g3 = {0, 0, 0, 0};
#if __clang_major__ >= 23
  v8i g4 = {0, 0, 0, 0, 0, 0, 0, 0};
  __builtin_amdgcn_tensor_load_to_lds(g0, g1, g2, g3, g4, 0);
#else
  __builtin_amdgcn_tensor_load_to_lds(g0, g1, g2, g3, 0);
#endif
}
#endif  // HAVE_TDM

// ---------------------------------------------------------------------------
// GEMM: Y[M x N] = X[M x K] @ W[N x K]^T + bias[N]   (fp32 in/out, bf16 WMMA)
// Block = 256 thr (8 waves); each wave owns one 16x16 output tile.
// Operands stream from global through the 192MB L2 (whole problem is L2-resident).
// ---------------------------------------------------------------------------
__global__ void __launch_bounds__(256) gemm_xwt_bf16(
    const float* __restrict__ X, const float* __restrict__ W,
    const float* __restrict__ bias, float* __restrict__ Y,
    int Mr, int Nc, int Kd) {
  const int tid = threadIdx.x;
  const int lane = tid & 31;
  const int wv = tid >> 5;
  const int row0 = blockIdx.y * 32 + (wv & 1) * 16;
  const int col0 = blockIdx.x * 64 + (wv >> 1) * 16;

  v8f acc = {0.f, 0.f, 0.f, 0.f, 0.f, 0.f, 0.f, 0.f};
  const int r = row0 + (lane & 15);
  const int abase = (lane >> 4) << 4;
  const int n = col0 + (lane & 15);

  for (int k0 = 0; k0 < Kd; k0 += 32) {
    if (k0 + 32 < Kd) {  // prefetch next K tile (global_prefetch_b8)
      __builtin_prefetch(X + (size_t)r * Kd + k0 + 32 + abase, 0, 1);
      __builtin_prefetch(W + (size_t)n * Kd + k0 + 32, 0, 1);
    }
    v16bf a, b;
#pragma unroll
    for (int e = 0; e < 16; ++e)
      a[e] = (__bf16)X[(size_t)r * Kd + k0 + abase + e];
    const int koff = k0 + ((lane >> 4) << 4);
#pragma unroll
    for (int e = 0; e < 16; ++e)
      b[e] = (__bf16)W[(size_t)n * Kd + koff + e];
    acc = wmma_bf16(a, b, acc);
  }

  const int rb = (lane >> 4) << 3;
  const int cb = lane & 15;
#pragma unroll
  for (int i = 0; i < 8; ++i) {
    const int rr = row0 + i + rb;
    const int cc = col0 + cb;
    Y[(size_t)rr * Nc + cc] = acc[i] + bias[cc];
  }
}

// ---------------------------------------------------------------------------
// Per-token feature map phi[p*4+m] = (x_n . poly_proj[:,p])^2 * prf_m,
// prf_m = 0.5*exp(clip((x_n . omega[:,m])*sqrt(2s) - s, +-20)),  s = 1/(2+1e-6).
// Values reach ~2.4e8 -> bf16 mandatory (FP16 max 65504 would overflow).
// ---------------------------------------------------------------------------
static __device__ inline void compute_phi(const float* __restrict__ row,
                                          const float* __restrict__ pp,
                                          const float* __restrict__ om,
                                          __bf16* __restrict__ dst,
                                          float sqrt2s, float s_node) {
  float ss = 0.f;
#pragma unroll 8
  for (int d = 0; d < HD; ++d) { const float v = row[d]; ss += v * v; }
  const float invn = 1.0f / fmaxf(sqrtf(ss), 1e-12f);

  float poly[PP];
  float prj[MMF];
#pragma unroll
  for (int p = 0; p < PP; ++p) poly[p] = 0.f;
#pragma unroll
  for (int m = 0; m < MMF; ++m) prj[m] = 0.f;

  for (int d = 0; d < HD; ++d) {
    const float v = row[d] * invn;
    const float* pr = pp + d * PP;
#pragma unroll
    for (int p = 0; p < PP; ++p) poly[p] = fmaf(v, pr[p], poly[p]);
    const float* orow = om + d * MMF;
#pragma unroll
    for (int m = 0; m < MMF; ++m) prj[m] = fmaf(v, orow[m], prj[m]);
  }
  float prf[MMF];
#pragma unroll
  for (int m = 0; m < MMF; ++m) {
    const float e = fminf(fmaxf(prj[m] * sqrt2s - s_node, -20.f), 20.f);
    prf[m] = 0.5f * expf(e);  // 1/sqrt(M)=0.5
  }
#pragma unroll
  for (int p = 0; p < PP; ++p) {
    const float pq = poly[p] * poly[p];
#pragma unroll
    for (int m = 0; m < MMF; ++m) dst[p * MMF + m] = (__bf16)(pq * prf[m]);
  }
}

// ---------------------------------------------------------------------------
// Chunked causal linear attention, one workgroup per head.
// State S[128x64] lives in persistent fp32 WMMA accumulators (4 x v8f per wave,
// 8 waves = whole S); a bf16 mirror in LDS feeds the phi_q @ S GEMM.
// V chunks are DMA'd global->LDS by the Tensor Data Mover (TENSORcnt).
// ---------------------------------------------------------------------------
__global__ void __launch_bounds__(256) yat_attn(
    const float* __restrict__ qkv, const float* __restrict__ omega,
    const float* __restrict__ poly_proj, float* __restrict__ attn) {
  __shared__ __attribute__((aligned(16))) __bf16 sPhiQ[CH][FD];  // 8 KB
  __shared__ __attribute__((aligned(16))) __bf16 sPhiK[CH][FD];  // 8 KB
  __shared__ __attribute__((aligned(16))) float  sV32[CH][HD];   // 8 KB (TDM dest)
  __shared__ __attribute__((aligned(16))) __bf16 sA[CH][CH];     // 2 KB
  __shared__ __attribute__((aligned(16))) __bf16 sS[FD][HD];     // 16 KB
  __shared__ float sZ[FD];   // running sum of phi_k
  __shared__ float sNorm[CH];

  const int h = blockIdx.x;
  const int tid = threadIdx.x;
  const int lane = tid & 31;
  const int wv = tid >> 5;

  const float s_node = 1.0f / (2.0f + 1e-6f);  // Gauss-Laguerre node (R=1) / C
  const float sqrt2s = sqrtf(2.0f * s_node);

  for (int i = tid; i < FD * HD; i += 256) ((__bf16*)sS)[i] = (__bf16)0.0f;
  if (tid < FD) sZ[tid] = 0.0f;

  const v8f z8 = {0.f, 0.f, 0.f, 0.f, 0.f, 0.f, 0.f, 0.f};
  v8f Sacc[4];
#pragma unroll
  for (int j = 0; j < 4; ++j) Sacc[j] = z8;

  __syncthreads();

  const int rb = (lane >> 4) << 3;
  const int cb = lane & 15;
  const int cti = wv >> 2, ctj = wv & 3;  // this wave's 16x16 ctx tile of [32x64]
  const int ati = wv >> 1, atj = wv & 1;  // this wave's A tile (wv<4) of [32x32]

  for (int chn = 0; chn < T_LEN / CH; ++chn) {
    const int t0 = chn * CH;

    // ---- stage 1: features (VALU on waves 0-1); V chunk via TDM (wave 7) ----
    if (tid < CH) {
      const float* row = qkv + (size_t)(t0 + tid) * (3 * EMBED) + h * HD;  // q
      compute_phi(row, poly_proj + h * HD * PP, omega + h * HD * MMF,
                  &sPhiQ[tid][0], sqrt2s, s_node);
    } else if (tid < 2 * CH) {
      const int tl = tid - CH;
      const float* row = qkv + (size_t)(t0 + tl) * (3 * EMBED) + EMBED + h * HD;
      compute_phi(row, poly_proj + h * HD * PP, omega + h * HD * MMF,
                  &sPhiK[tl][0], sqrt2s, s_node);
    }
#if HAVE_TDM
    else if (wv == 7) {
      const unsigned long long gaddr = (unsigned long long)(uintptr_t)(
          qkv + (size_t)t0 * (3 * EMBED) + 2 * EMBED + h * HD);
      tdm_load_2d_f32(gaddr, (unsigned)(uintptr_t)&sV32[0][0], HD, CH,
                      (unsigned long long)(3 * EMBED));
      __builtin_amdgcn_s_wait_tensorcnt(0);
    }
#else
    else {
      for (int i = tid - 2 * CH; i < CH * HD; i += 256 - 2 * CH)
        sV32[i >> 6][i & 63] =
            qkv[(size_t)(t0 + (i >> 6)) * (3 * EMBED) + 2 * EMBED + h * HD +
                (i & 63)];
    }
#endif
    __syncthreads();

    // ---- stage 2: norm base, ctx_inter = PhiQ @ S, A = PhiQ @ PhiK^T --------
    if (tid < CH) {
      float acc = 0.f;
      for (int f = 0; f < FD; ++f) acc += (float)sPhiQ[tid][f] * sZ[f];
      sNorm[tid] = acc;
    }
    v8f ctx = z8;
#pragma unroll
    for (int kk = 0; kk < FD; kk += 32)
      ctx = wmma_bf16(load_a_rm(&sPhiQ[0][0], FD, cti * 16, kk, lane),
                      load_b_rm(&sS[0][0], HD, kk, ctj * 16, lane), ctx);
    v8f areg = z8;
    if (wv < 4) {
#pragma unroll
      for (int kk = 0; kk < FD; kk += 32)
        areg = wmma_bf16(load_a_rm(&sPhiQ[0][0], FD, ati * 16, kk, lane),
                         load_b_tr(&sPhiK[0][0], FD, kk, atj * 16, lane), areg);
    }
    __syncthreads();

    // ---- stage 3: causal mask + store A (bf16) ------------------------------
    if (wv < 4) {
#pragma unroll
      for (int i = 0; i < 8; ++i) {
        const int rr = ati * 16 + i + rb;
        const int cc = atj * 16 + cb;
        sA[rr][cc] = (__bf16)((cc <= rr) ? areg[i] : 0.0f);
      }
    }
    __syncthreads();

    // ---- stage 4: ctx += A@V ; S += PhiK^T @ V ; norm rowsums ; z update ----
    ctx = wmma_bf16(load_a_rm(&sA[0][0], CH, cti * 16, 0, lane),
                    load_b_rm_f32(&sV32[0][0], HD, 0, ctj * 16, lane), ctx);
    {
      const v16bf akt = load_a_tr(&sPhiK[0][0], FD, wv * 16, 0, lane);
#pragma unroll
      for (int dj = 0; dj < 4; ++dj)
        Sacc[dj] = wmma_bf16(akt, load_b_rm_f32(&sV32[0][0], HD, 0, dj * 16, lane),
                             Sacc[dj]);
    }
    if (tid < CH) {
      float acc = sNorm[tid];
      for (int c = 0; c <= tid; ++c) acc += (float)sA[tid][c];
      sNorm[tid] = acc;
    } else if (tid < CH + FD) {
      const int f = tid - CH;
      float acc = 0.f;
      for (int j = 0; j < CH; ++j) acc += (float)sPhiK[j][f];
      sZ[f] += acc;
    }
    __syncthreads();

    // ---- stage 5: write out ctx/norm ; refresh bf16 mirror of S -------------
#pragma unroll
    for (int i = 0; i < 8; ++i) {
      const int rr = cti * 16 + i + rb;
      const float nv = fmaxf(sNorm[rr], 1e-6f);
      attn[(size_t)(t0 + rr) * EMBED + h * HD + ctj * 16 + cb] = ctx[i] / nv;
    }
#pragma unroll
    for (int dj = 0; dj < 4; ++dj)
#pragma unroll
      for (int i = 0; i < 8; ++i)
        sS[wv * 16 + i + rb][dj * 16 + cb] = (__bf16)Sacc[dj][i];
    __syncthreads();
  }
}

// ---------------------------------------------------------------------------
extern "C" void kernel_launch(void* const* d_in, const int* in_sizes, int n_in,
                              void* d_out, int out_size, void* d_ws, size_t ws_size,
                              hipStream_t stream) {
  const float* x         = (const float*)d_in[0];
  const float* qkv_w     = (const float*)d_in[1];
  const float* qkv_b     = (const float*)d_in[2];
  const float* out_w     = (const float*)d_in[3];
  const float* out_b     = (const float*)d_in[4];
  const float* omega     = (const float*)d_in[5];
  const float* poly_proj = (const float*)d_in[6];
  float* outp = (float*)d_out;

  float* qkv  = (float*)d_ws;                         // [1024 x 1536] fp32
  float* attn = qkv + (size_t)T_LEN * (3 * EMBED);    // [1024 x 512]  fp32

  // 1) qkv = x @ qkv_w^T + qkv_b
  dim3 g1((3 * EMBED) / 64, T_LEN / 32);
  gemm_xwt_bf16<<<g1, 256, 0, stream>>>(x, qkv_w, qkv_b, qkv, T_LEN, 3 * EMBED,
                                        EMBED);
  // 2) chunked causal linear attention, one WG per head
  yat_attn<<<dim3(NH), 256, 0, stream>>>(qkv, omega, poly_proj, attn);
  // 3) out = attn @ out_w^T + out_b
  dim3 g3(EMBED / 64, T_LEN / 32);
  gemm_xwt_bf16<<<g3, 256, 0, stream>>>(attn, out_w, out_b, outp, T_LEN, EMBED,
                                        EMBED);
}
